// HeteroGAT_4114578669779
// MI455X (gfx1250) — matrix operand order repaired and checked by reading, not compiled, not run on hardware
//
#include <hip/hip_runtime.h>

#define CH 128
#define NHEADS 4
#define HDIM 32

typedef __attribute__((ext_vector_type(16))) __bf16 v16bf;
typedef __attribute__((ext_vector_type(8)))  __bf16 v8bf;
typedef __attribute__((ext_vector_type(4)))  __bf16 v4bf;
typedef __attribute__((ext_vector_type(8)))  float  v8f;

// ---- monotonic float <-> ordered int mapping (for atomicMax on float) ----
__device__ __forceinline__ int ord_from_f32(float f) {
  int i = __float_as_int(f);
  return (i >= 0) ? i : (i ^ 0x7FFFFFFF);
}
__device__ __forceinline__ float f32_from_ord(int i) {
  return __int_as_float((i >= 0) ? i : (i ^ 0x7FFFFFFF));
}
#define ORD_NEG_INF ((int)0x807FFFFF)  // ord_from_f32(-inf)

// ============================================================================
// fp32 -> bf16 conversion, 4 elements/thread (n must be a multiple of 4)
// ============================================================================
__global__ __launch_bounds__(256) void cvt_bf16_k(
    const float* __restrict__ X, v4bf* __restrict__ Y, long long n4) {
  long long t = (long long)blockIdx.x * blockDim.x + threadIdx.x;
  if (t >= n4) return;
  float4 x = ((const float4*)X)[t];
  v4bf y;
  y[0] = (__bf16)x.x; y[1] = (__bf16)x.y; y[2] = (__bf16)x.z; y[3] = (__bf16)x.w;
  Y[t] = y;
}

// ============================================================================
// Bt[n*K + k] = (bf16) W[k*N + n]   (transpose + convert; W is tiny, L2-hot)
// ============================================================================
__global__ void cvt_transpose_bf16_k(const float* __restrict__ W,
                                     __bf16* __restrict__ Bt, int K, int N) {
  int t = blockIdx.x * blockDim.x + threadIdx.x;
  if (t >= K * N) return;
  int n = t / K, k = t - n * K;
  Bt[t] = (__bf16)W[(size_t)k * N + n];
}

// ============================================================================
// C[M,N] = A[M,128] @ B[128,N] (+ bias[N]); A bf16 row-major, Bt = B^T bf16.
// One block per 16-row M tile; each wave owns one 16-col N tile.
// K fixed = 128. Per tile: clause of 8x global_load_b128 (B strip), then
// 4x v_wmma back-to-back with partial load waits (software pipelined).
// A operand assembled once per wave, register-resident across tiles.
// ============================================================================
__global__ __launch_bounds__(256) void wmma_gemm_bf16(
    const __bf16* __restrict__ A,   // [M, 128] row-major
    const __bf16* __restrict__ Bt,  // [N, 128] (= B transposed)
    const float* __restrict__ bias, float* __restrict__ C,
    int M, int N) {
  const int mtile  = blockIdx.x;
  const int wave   = threadIdx.x >> 5;
  const int nwaves = blockDim.x >> 5;
  const int lane   = threadIdx.x & 31;
  const int r16    = lane & 15;
  const int hi     = lane >> 4;  // 0: lanes 0-15, 1: lanes 16-31
  const int ntiles = N >> 4;

  // ---- A operand: row M = lane&15; lanes 0-15 hold K {0..7,16..23}+k0,
  // lanes 16-31 hold K {8..15,24..31}+k0 (ISA 16-bit A 16x32 layout). ----
  const __bf16* Arow = A + (size_t)(mtile * 16 + r16) * CH;
  v16bf av[4];
  #pragma unroll
  for (int j = 0; j < 4; ++j) {
    v8bf alo = *(const v8bf*)(Arow + j * 32 + hi * 8);
    v8bf ahi = *(const v8bf*)(Arow + j * 32 + hi * 8 + 16);
    av[j] = __builtin_shufflevector(alo, ahi, 0, 1, 2, 3, 4, 5, 6, 7,
                                    8, 9, 10, 11, 12, 13, 14, 15);
  }

  for (int nt = wave; nt < ntiles; nt += nwaves) {
    // ---- B operand: column N = lane&15; lanes 0-15 hold K {0..7,16..23}+k0,
    // lanes 16-31 hold K {8..15,24..31}+k0 (same K order as A). ----
    const __bf16* Brow = Bt + (size_t)(nt * 16 + r16) * CH;
    v8bf b0[4], b1[4];
    #pragma unroll
    for (int j = 0; j < 4; ++j) {       // issue all 8 loads first (one clause)
      b0[j] = *(const v8bf*)(Brow + j * 32 + hi * 8);
      b1[j] = *(const v8bf*)(Brow + j * 32 + hi * 8 + 16);
    }
    v8f acc = {};
    #pragma unroll
    for (int j = 0; j < 4; ++j) {       // then the WMMA chain
      v16bf bv = __builtin_shufflevector(b0[j], b1[j], 0, 1, 2, 3, 4, 5, 6, 7,
                                         8, 9, 10, 11, 12, 13, 14, 15);
      acc = __builtin_amdgcn_wmma_f32_16x16x32_bf16(
          false, av[j], false, bv, (short)0, acc, false, false);
    }
    const int col = nt * 16 + r16;
    const float badd = bias ? bias[col] : 0.0f;
    #pragma unroll
    for (int r = 0; r < 8; ++r) {  // C/D: VGPR r -> row r + 8*hi
      C[(size_t)(mtile * 16 + hi * 8 + r) * N + col] = acc[r] + badd;
    }
  }
}

// ============================================================================
// es[n,h] = sum_c hs[n, h*32+c] * a_s[h*32+c]   (one wave32 per node)
// ============================================================================
__global__ __launch_bounds__(256) void src_logits_k(
    const float* __restrict__ HS, const float* __restrict__ a_s,
    float* __restrict__ ES, int n) {
  int node = blockIdx.x * (blockDim.x >> 5) + (threadIdx.x >> 5);
  int lane = threadIdx.x & 31;
  if (node >= n) return;
  const float* row = HS + (size_t)node * CH;
  float p0 = row[lane]      * a_s[lane];
  float p1 = row[32 + lane] * a_s[32 + lane];
  float p2 = row[64 + lane] * a_s[64 + lane];
  float p3 = row[96 + lane] * a_s[96 + lane];
  #pragma unroll
  for (int off = 16; off > 0; off >>= 1) {
    p0 += __shfl_xor(p0, off, 32);
    p1 += __shfl_xor(p1, off, 32);
    p2 += __shfl_xor(p2, off, 32);
    p3 += __shfl_xor(p3, off, 32);
  }
  if (lane == 0) {
    float* o = ES + (size_t)node * NHEADS;
    o[0] = p0; o[1] = p1; o[2] = p2; o[3] = p3;
  }
}

// ============================================================================
// Fold Wd with a_d: VF[k*4+h] = sum_c Wd[k,h*32+c] * a_d[h,c]
// ============================================================================
__global__ void fold_dst_k(const float* __restrict__ Wd,
                           const float* __restrict__ a_d,
                           float* __restrict__ VF) {
  int t = blockIdx.x * blockDim.x + threadIdx.x;
  if (t >= CH * NHEADS) return;
  int k = t >> 2, h = t & 3;
  float s = 0.f;
  #pragma unroll
  for (int c = 0; c < HDIM; ++c)
    s += Wd[(size_t)k * CH + h * HDIM + c] * a_d[h * HDIM + c];
  VF[t] = s;  // layout [k][h], float4-addressable per k
}

// ============================================================================
// ed[n,h] = sum_k x_dst[n,k] * VF[k,h]   (one wave32 per node)
// ============================================================================
__global__ __launch_bounds__(256) void dst_logits_k(
    const float* __restrict__ X, const float* __restrict__ VF,
    float* __restrict__ ED, int n) {
  int node = blockIdx.x * (blockDim.x >> 5) + (threadIdx.x >> 5);
  int lane = threadIdx.x & 31;
  if (node >= n) return;
  const float* row = X + (size_t)node * CH;
  const float4* vf = (const float4*)VF;
  float p0 = 0.f, p1 = 0.f, p2 = 0.f, p3 = 0.f;
  #pragma unroll
  for (int j = 0; j < 4; ++j) {
    int k = j * 32 + lane;
    float xv = row[k];
    float4 v = vf[k];
    p0 += xv * v.x; p1 += xv * v.y; p2 += xv * v.z; p3 += xv * v.w;
  }
  #pragma unroll
  for (int off = 16; off > 0; off >>= 1) {
    p0 += __shfl_xor(p0, off, 32);
    p1 += __shfl_xor(p1, off, 32);
    p2 += __shfl_xor(p2, off, 32);
    p3 += __shfl_xor(p3, off, 32);
  }
  if (lane == 0) {
    float* o = ED + (size_t)node * NHEADS;
    o[0] = p0; o[1] = p1; o[2] = p2; o[3] = p3;
  }
}

// ============================================================================
// Zero aggregation buffer, z-sums; set segment-max to ordered(-inf)
// ============================================================================
__global__ void init_dir_k(int* __restrict__ Mb, float* __restrict__ Z,
                           float* __restrict__ AGG, int ndst) {
  long long t = (long long)blockIdx.x * blockDim.x + threadIdx.x;
  long long nh = (long long)ndst * NHEADS;
  if (t < nh) { Mb[t] = ORD_NEG_INF; Z[t] = 0.0f; }
  long long nc = (long long)ndst * CH;
  if (t < nc) AGG[t] = 0.0f;
}

// ============================================================================
// Pass 1: e = leaky_relu(es[src]+ed[dst]); segment max via ordered-int atomicMax
// ============================================================================
__global__ void edge_max_k(const int* __restrict__ src, const int* __restrict__ dst,
                           const float* __restrict__ ES, const float* __restrict__ ED,
                           float* __restrict__ WE, int* __restrict__ Mb, int E) {
  long long t = (long long)blockIdx.x * blockDim.x + threadIdx.x;
  if (t >= (long long)E * NHEADS) return;
  int e = (int)(t >> 2), h = (int)(t & 3);
  int s = src[e], d = dst[e];
  float v = ES[(size_t)s * NHEADS + h] + ED[(size_t)d * NHEADS + h];
  v = (v > 0.f) ? v : v * 0.2f;
  WE[t] = v;
  atomicMax(&Mb[(size_t)d * NHEADS + h], ord_from_f32(v));
}

// ============================================================================
// Pass 2: w = exp(e - m[dst]); segment sum z += w  (w overwrites e in place)
// ============================================================================
__global__ void edge_expsum_k(const int* __restrict__ dst, float* __restrict__ WE,
                              const int* __restrict__ Mb, float* __restrict__ Z, int E) {
  long long t = (long long)blockIdx.x * blockDim.x + threadIdx.x;
  if (t >= (long long)E * NHEADS) return;
  int e = (int)(t >> 2), h = (int)(t & 3);
  int d = dst[e];
  float m = f32_from_ord(Mb[(size_t)d * NHEADS + h]);
  float w = expf(WE[t] - m);
  WE[t] = w;
  atomicAdd(&Z[(size_t)d * NHEADS + h], w);
}

// ============================================================================
// Pass 3: agg[dst,c] += hs[src,c] * (w / (z[dst,h]+1e-16)); 128 lanes per edge
// ============================================================================
__global__ void edge_agg_k(const int* __restrict__ src, const int* __restrict__ dst,
                           const float* __restrict__ WE, const float* __restrict__ Z,
                           const float* __restrict__ HS, float* __restrict__ AGG, int E) {
  long long t = (long long)blockIdx.x * blockDim.x + threadIdx.x;
  if (t >= (long long)E * CH) return;
  int e = (int)(t >> 7);
  int c = (int)(t & 127);
  int h = c >> 5;
  int s = src[e], d = dst[e];
  float alpha = WE[(size_t)e * NHEADS + h] / (Z[(size_t)d * NHEADS + h] + 1e-16f);
  atomicAdd(&AGG[(size_t)d * CH + c], HS[(size_t)s * CH + c] * alpha);
}

// ============================================================================
// In-place: x = relu(layernorm(agg + bias) * lw + lb)  (one wave32 per node)
// ============================================================================
__global__ __launch_bounds__(256) void ln_relu_k(
    float* __restrict__ X, const float* __restrict__ bias,
    const float* __restrict__ lw, const float* __restrict__ lb, int n) {
  int node = blockIdx.x * (blockDim.x >> 5) + (threadIdx.x >> 5);
  int lane = threadIdx.x & 31;
  if (node >= n) return;
  float* row = X + (size_t)node * CH;
  float v[4]; float s = 0.f, sq = 0.f;
  #pragma unroll
  for (int j = 0; j < 4; ++j) {
    int c = j * 32 + lane;
    float x = row[c] + bias[c];
    v[j] = x; s += x; sq += x * x;
  }
  #pragma unroll
  for (int off = 16; off > 0; off >>= 1) {
    s  += __shfl_xor(s, off, 32);
    sq += __shfl_xor(sq, off, 32);
  }
  float mu  = s * (1.0f / CH);
  float var = sq * (1.0f / CH) - mu * mu;   // biased, matches jnp.var
  float inv = rsqrtf(var + 1e-5f);
  #pragma unroll
  for (int j = 0; j < 4; ++j) {
    int c = j * 32 + lane;
    float y = (v[j] - mu) * inv * lw[c] + lb[c];
    row[c] = fmaxf(y, 0.f);
  }
}

// ============================================================================
// Host-side orchestration
// ============================================================================
static inline unsigned cdiv_u(long long a, long long b) {
  return (unsigned)((a + b - 1) / b);
}

static void run_direction(hipStream_t stream,
                          const float* xsrc, int nsrc,
                          const float* xdst, int ndst,
                          const int* src, const int* dst, int E,
                          const float* Wsrc, const float* Wdst,
                          const float* a_s, const float* a_d,
                          __bf16* Abf, __bf16* Btb,
                          float* HS, float* ES, float* ED, float* VF,
                          float* Z, int* Mb, float* WE, float* AGG) {
  // bf16 staging: A = x_src (row-major), Bt = Wsrc^T (column-major)
  long long n4 = (long long)nsrc * CH / 4;
  cvt_bf16_k<<<cdiv_u(n4, 256), 256, 0, stream>>>(xsrc, (v4bf*)Abf, n4);
  cvt_transpose_bf16_k<<<cdiv_u(CH * CH, 256), 256, 0, stream>>>(Wsrc, Btb, CH, CH);
  // hs = x_src @ Wsrc   (WMMA, packed bf16 operands)
  wmma_gemm_bf16<<<nsrc / 16, 256, 0, stream>>>(Abf, Btb, nullptr, HS, nsrc, CH);
  // per-node logits
  src_logits_k<<<cdiv_u(nsrc, 8), 256, 0, stream>>>(HS, a_s, ES, nsrc);
  fold_dst_k<<<2, 256, 0, stream>>>(Wdst, a_d, VF);
  dst_logits_k<<<cdiv_u(ndst, 8), 256, 0, stream>>>(xdst, VF, ED, ndst);
  // segment softmax + weighted aggregation
  init_dir_k<<<cdiv_u((long long)ndst * CH, 256), 256, 0, stream>>>(Mb, Z, AGG, ndst);
  edge_max_k<<<cdiv_u((long long)E * NHEADS, 256), 256, 0, stream>>>(src, dst, ES, ED, WE, Mb, E);
  edge_expsum_k<<<cdiv_u((long long)E * NHEADS, 256), 256, 0, stream>>>(dst, WE, Mb, Z, E);
  edge_agg_k<<<cdiv_u((long long)E * CH, 256), 256, 0, stream>>>(src, dst, WE, Z, HS, AGG, E);
}

extern "C" void kernel_launch(void* const* d_in, const int* in_sizes, int n_in,
                              void* d_out, int out_size, void* d_ws, size_t ws_size,
                              hipStream_t stream) {
  (void)in_sizes; (void)n_in; (void)out_size; (void)ws_size;

  const float* x_user  = (const float*)d_in[0];
  const float* x_item  = (const float*)d_in[1];
  const int*   ui_src  = (const int*)d_in[2];
  const int*   ui_dst  = (const int*)d_in[3];
  const int*   iu_src  = (const int*)d_in[4];
  const int*   iu_dst  = (const int*)d_in[5];
  const float* Wsrc_ui = (const float*)d_in[6];
  const float* Wdst_ui = (const float*)d_in[7];
  const float* asrc_ui = (const float*)d_in[8];
  const float* adst_ui = (const float*)d_in[9];
  const float* b_ui    = (const float*)d_in[10];
  const float* Wsrc_iu = (const float*)d_in[11];
  const float* Wdst_iu = (const float*)d_in[12];
  const float* asrc_iu = (const float*)d_in[13];
  const float* adst_iu = (const float*)d_in[14];
  const float* b_iu    = (const float*)d_in[15];
  const float* ln_w_u  = (const float*)d_in[16];
  const float* ln_b_u  = (const float*)d_in[17];
  const float* ln_w_i  = (const float*)d_in[18];
  const float* ln_b_i  = (const float*)d_in[19];
  const float* lin_w   = (const float*)d_in[20];
  const float* lin_b   = (const float*)d_in[21];

  const int NU = 100000, NI = 200000, E = 500000, Brows = 1024, OUTD = 64;

  // --- workspace carve-out ---
  float* ws = (float*)d_ws;
  size_t off = 0;  // in fp32 elements; every block below is 16B-aligned
  float* U[2]; float* I[2];
  U[0] = ws + off; off += (size_t)NU * CH;
  U[1] = ws + off; off += (size_t)NU * CH;
  I[0] = ws + off; off += (size_t)NI * CH;
  I[1] = ws + off; off += (size_t)NI * CH;
  float* HS = ws + off; off += (size_t)NI * CH;       // transformed src feats
  float* ES = ws + off; off += (size_t)NI * NHEADS;   // src logits
  float* ED = ws + off; off += (size_t)NI * NHEADS;   // dst logits
  float* VF = ws + off; off += (size_t)CH * NHEADS;   // folded Wd*a_d
  float* Z  = ws + off; off += (size_t)NI * NHEADS;   // softmax denominators
  int*   Mb = (int*)(ws + off); off += (size_t)NI * NHEADS; // segment max (ordered)
  float* WE = ws + off; off += (size_t)E * NHEADS;    // per-edge logits -> weights
  __bf16* Abf = (__bf16*)(ws + off); off += (size_t)NI * CH / 2;  // bf16 A staging
  __bf16* Btb = (__bf16*)(ws + off); off += (size_t)CH * CH / 2;  // bf16 B^T staging

  const float* cu = x_user;
  const float* ci = x_item;
  for (int l = 0; l < 2; ++l) {
    const size_t wo = (size_t)l * CH * CH;
    const size_t ao = (size_t)l * NHEADS * HDIM;
    const size_t co = (size_t)l * CH;
    // user update: aggregate items -> users over iu edges
    run_direction(stream, ci, NI, cu, NU, iu_src, iu_dst, E,
                  Wsrc_iu + wo, Wdst_iu + wo, asrc_iu + ao, adst_iu + ao,
                  Abf, Btb, HS, ES, ED, VF, Z, Mb, WE, U[l]);
    // item update: aggregate users -> items over ui edges
    run_direction(stream, cu, NU, ci, NI, ui_src, ui_dst, E,
                  Wsrc_ui + wo, Wdst_ui + wo, asrc_ui + ao, adst_ui + ao,
                  Abf, Btb, HS, ES, ED, VF, Z, Mb, WE, I[l]);
    // x = relu(LN(agg + b)) in place
    ln_relu_k<<<cdiv_u(NU, 8), 256, 0, stream>>>(U[l], b_iu + co, ln_w_u + co, ln_b_u + co, NU);
    ln_relu_k<<<cdiv_u(NI, 8), 256, 0, stream>>>(I[l], b_ui + co, ln_w_i + co, ln_b_i + co, NI);
    cu = U[l]; ci = I[l];
  }
  // out = x_user[:1024] @ lin_w + lin_b   (WMMA)
  long long n4 = (long long)Brows * CH / 4;
  cvt_bf16_k<<<cdiv_u(n4, 256), 256, 0, stream>>>(cu, (v4bf*)Abf, n4);
  cvt_transpose_bf16_k<<<cdiv_u(CH * OUTD, 256), 256, 0, stream>>>(lin_w, Btb, CH, OUTD);
  wmma_gemm_bf16<<<Brows / 16, 256, 0, stream>>>(Abf, Btb, lin_b, (float*)d_out,
                                                 Brows, OUTD);
}
